// HyenaFilter_14972255994528
// MI455X (gfx1250) — matrix operand order repaired and checked by reading, not compiled, not run on hardware
//
#include <hip/hip_runtime.h>
#include <hip/hip_bf16.h>

typedef __attribute__((ext_vector_type(16))) _Float16 v16h;
typedef __attribute__((ext_vector_type(8)))  float    v8f;

#define LSEQ   8192
#define NFFT   16384
#define DMODEL 768
#define EMB    33
#define FO     64
#define KHAT_STRIDE 8256      // >= 8193, padded
#define K1_THREADS 128
#define K1_WAVES   4

// fragment slot sizes (in halfs)
#define FRAG   16                       // 16 halfs per lane per fragment
#define BSLOT  (32 * FRAG)              // one (kt,nt) B-fragment: 32 lanes
#define ASLOT  (32 * FRAG)              // one kt A-fragment: 32 lanes

// base-4 digit reversal of a 14-bit (7-digit) index
__device__ __forceinline__ int drev14(int v) {
    int r = 0;
#pragma unroll
    for (int i = 0; i < 7; ++i) { r = (r << 2) | (v & 3); v >>= 2; }
    return r;
}

__device__ __forceinline__ float2 cmul2(float2 x, float cr, float ci) {
    return make_float2(x.x * cr - x.y * ci, x.x * ci + x.y * cr);
}

// Write activation value for K-column c of the next layer's A matrix into the
// fragment-native A buffer: A[(kt*32 + row + 16*half)*16 + e]
__device__ __forceinline__ void a_scatter(_Float16* Asw, int row, int c, float val)
{
    int kt  = c >> 5;
    int c32 = c & 31;
    int e   = (c32 & 7) + ((c32 & 16) ? 8 : 0);
    int hp  = (c32 >> 3) & 1;
    Asw[(kt * 32 + row + 16 * hp) * FRAG + e] = (_Float16)val;
}

// ---------------------------------------------------------------------------
// 16x64 @ 64x(16*ntiles) via v_wmma_f32_16x16x32_f16 with fragment-native LDS.
// Asw: [2][32][16] halfs (this wave).  Bsw: [2][ntiles][32][16] halfs (shared).
// Epilogue: sin(freq*(.+bias)) scattered into AswOut fragment layout.
// ---------------------------------------------------------------------------
__device__ __forceinline__ void dense_frag(const _Float16* __restrict__ Asw,
                                           _Float16* __restrict__ AswOut,
                                           const _Float16* __restrict__ Bsw,
                                           const float* __restrict__ bias,
                                           const float* __restrict__ freq,
                                           int lane)
{
    const int M  = lane & 15;
    const int hi = (lane >> 4) & 1;
    const v16h a0 = *(const v16h*)(Asw + (0 * 32 + lane) * FRAG);
    const v16h a1 = *(const v16h*)(Asw + (1 * 32 + lane) * FRAG);
#pragma unroll
    for (int nt = 0; nt < 4; ++nt) {
        v8f acc = {0.f,0.f,0.f,0.f,0.f,0.f,0.f,0.f};
        v16h b0 = *(const v16h*)(Bsw + ((0 * 4 + nt) * 32 + lane) * FRAG);
        acc = __builtin_amdgcn_wmma_f32_16x16x32_f16(false, a0, false, b0,
                                                     (short)0, acc, false, false);
        v16h b1 = *(const v16h*)(Bsw + ((1 * 4 + nt) * 32 + lane) * FRAG);
        acc = __builtin_amdgcn_wmma_f32_16x16x32_f16(false, a1, false, b1,
                                                     (short)0, acc, false, false);
        int col = nt * 16 + M;
        float fq = freq[col], bs = bias[col];
#pragma unroll
        for (int j = 0; j < 8; ++j) {
            int row = j + hi * 8;           // C/D: VGPR j -> M = j + 8*hi
            a_scatter(AswOut, row, col, __sinf(fq * (acc[j] + bs)));
        }
    }
}

// ---------------------------------------------------------------------------
// Kernel 1: implicit filter MLP -> k (DMODEL x LSEQ) in workspace.
// 128 threads = 4 waves, each owning one 16-row L-tile.  All weights staged
// once into fragment-native f16 LDS; zero padding baked in at staging.
// ---------------------------------------------------------------------------
__global__ void __launch_bounds__(K1_THREADS)
filter_mlp_kernel(const float* __restrict__ z,
                  const float* __restrict__ t,
                  const float* __restrict__ freq,
                  const float* __restrict__ W1, const float* __restrict__ b1,
                  const float* __restrict__ W2, const float* __restrict__ b2,
                  const float* __restrict__ W3, const float* __restrict__ b3,
                  const float* __restrict__ Wout,
                  const float* __restrict__ deltas,
                  float* __restrict__ k_out)
{
    extern __shared__ unsigned char dyn_lds[];
    _Float16* Bsw1 = (_Float16*)dyn_lds;            // 2*4*32*16 = 4096 halfs
    _Float16* Bsw2 = Bsw1 + 2 * 4 * BSLOT;
    _Float16* Bsw3 = Bsw2 + 2 * 4 * BSLOT;
    _Float16* BswO = Bsw3 + 2 * 4 * BSLOT;          // 2*48*32*16 = 49152 halfs
    _Float16* AswA = BswO + 2 * 48 * BSLOT;         // 4 waves * 2*32*16
    _Float16* AswB = AswA + K1_WAVES * 2 * ASLOT;
    float*    tpose = (float*)(AswB + K1_WAVES * 2 * ASLOT); // 4 waves * 16*16

    const int tid  = threadIdx.x;
    const int wv   = tid >> 5;
    const int lane = tid & 31;
    const int M    = lane & 15;
    const int hi   = (lane >> 4) & 1;
    const int l0   = blockIdx.x * (K1_WAVES * 16) + wv * 16;

    // ---- stage 64x64 weights into fragment-native layout (zero padded) ----
    for (int i = tid; i < 2 * 4 * BSLOT; i += K1_THREADS) {
        int e  = i & 15;
        int ln = (i >> 4) & 31;
        int nt = (i >> 9) & 3;
        int kt = (i >> 11) & 1;
        int v = e >> 1, p = e & 1;
        int kb  = kt * 32 + ((2 * v + p) & 7) + (((2 * v + p) & 8) ? 16 : 0) + ((ln & 16) ? 8 : 0);
        int col = nt * 16 + (ln & 15);
        Bsw1[i] = (kb < EMB) ? (_Float16)W1[kb * FO + col] : (_Float16)0.0f;
        Bsw2[i] = (_Float16)W2[kb * FO + col];
        Bsw3[i] = (_Float16)W3[kb * FO + col];
    }
    // ---- stage Wout (64x768) into fragment-native layout ----
    for (int i = tid; i < 2 * 48 * BSLOT; i += K1_THREADS) {
        int e  = i & 15;
        int ln = (i >> 4) & 31;
        int nk = i >> 9;              // kt*48 + nt
        int nt = nk % 48;
        int kt = nk / 48;
        int v = e >> 1, p = e & 1;
        int kb  = kt * 32 + ((2 * v + p) & 7) + (((2 * v + p) & 8) ? 16 : 0) + ((ln & 16) ? 8 : 0);
        int col = nt * 16 + (ln & 15);
        BswO[i] = (_Float16)Wout[kb * DMODEL + col];
    }
    // ---- stage this wave's z tile straight into A-fragment layout ----
    _Float16* Aw = AswA + wv * 2 * ASLOT;
    _Float16* Bw = AswB + wv * 2 * ASLOT;
    for (int i = lane; i < 16 * FO; i += 32) {
        int r = i >> 6, c = i & 63;
        float val = (c < EMB) ? z[(size_t)(l0 + r) * EMB + c] : 0.0f;
        a_scatter(Aw, r, c, val);
    }
    __syncthreads();

    dense_frag(Aw, Bw, Bsw1, b1, freq, lane);   // layer 1: Aw -> Bw
    __syncthreads();
    dense_frag(Bw, Aw, Bsw2, b2, freq, lane);   // layer 2: Bw -> Aw
    __syncthreads();
    dense_frag(Aw, Bw, Bsw3, b3, freq, lane);   // layer 3: Aw -> Bw
    __syncthreads();

    // ---- output projection act @ Wout(64x768) + decay modulation ----
    const v16h a0 = *(const v16h*)(Bw + (0 * 32 + lane) * FRAG);
    const v16h a1 = *(const v16h*)(Bw + (1 * 32 + lane) * FRAG);
    float* tp = tpose + wv * 16 * 16;
    for (int nt = 0; nt < DMODEL / 16; ++nt) {
        v8f acc = {0.f,0.f,0.f,0.f,0.f,0.f,0.f,0.f};
        v16h bb0 = *(const v16h*)(BswO + ((0 * 48 + nt) * 32 + lane) * FRAG);
        acc = __builtin_amdgcn_wmma_f32_16x16x32_f16(false, a0, false, bb0,
                                                     (short)0, acc, false, false);
        v16h bb1 = *(const v16h*)(BswO + ((1 * 48 + nt) * 32 + lane) * FRAG);
        acc = __builtin_amdgcn_wmma_f32_16x16x32_f16(false, a1, false, bb1,
                                                     (short)0, acc, false, false);
        // D-tile round-trips through LDS unchanged; each lane writes 8 consecutive l of one channel
#pragma unroll
        for (int j = 0; j < 8; ++j)
            tp[(j + hi * 8) * 16 + M] = acc[j];
        __syncthreads();
        {
            int d = nt * 16 + M;
            float ad = fabsf(deltas[d]);
            int rbase = hi * 8;             // lane halves split the 16 rows
#pragma unroll
            for (int r = 0; r < 8; ++r) {
                int row = rbase + r, l = l0 + row;
                float dec = __expf(-t[l] * ad) + 0.05f;
                k_out[(size_t)d * LSEQ + l] = tp[row * 16 + M] * dec;
            }
        }
        __syncthreads();
    }
}

// ---------------------------------------------------------------------------
// Radix-4 in-LDS FFT, N = 16384 complex (float2), 256 threads.
// DIT: digit-reversed input -> natural output.
// DIF: natural input -> digit-reversed output.
// ---------------------------------------------------------------------------
__device__ void fft4_dit(float2* a, int tid)
{
    for (int s = 0; s < 7; ++s) {
        const int q  = 1 << (2 * s);
        const int m4 = q << 2;
        for (int j = tid; j < NFFT / 4; j += 256) {
            int grp = j >> (2 * s);
            int pos = j & (q - 1);
            int i0  = grp * m4 + pos;
            int i1 = i0 + q, i2 = i0 + 2 * q, i3 = i0 + 3 * q;
            float ang = -6.28318530717958647692f * (float)pos / (float)m4;
            float s1, c1; __sincosf(ang, &s1, &c1);
            float c2 = c1 * c1 - s1 * s1, s2 = 2.f * c1 * s1;
            float c3 = c2 * c1 - s2 * s1, s3 = c2 * s1 + s2 * c1;
            float2 t0 = a[i0];
            float2 t1 = cmul2(a[i1], c1, s1);
            float2 t2 = cmul2(a[i2], c2, s2);
            float2 t3 = cmul2(a[i3], c3, s3);
            float pr = t0.x + t2.x, pi = t0.y + t2.y;
            float ur = t0.x - t2.x, ui = t0.y - t2.y;
            float qr = t1.x + t3.x, qi = t1.y + t3.y;
            float vr = t1.x - t3.x, vi = t1.y - t3.y;
            a[i0] = make_float2(pr + qr, pi + qi);
            a[i2] = make_float2(pr - qr, pi - qi);
            a[i1] = make_float2(ur + vi, ui - vr);   // u + (-i)v
            a[i3] = make_float2(ur - vi, ui + vr);   // u + (+i)v
        }
        __syncthreads();
    }
}

__device__ void fft4_dif(float2* a, int tid)
{
    for (int s = 6; s >= 0; --s) {
        const int q  = 1 << (2 * s);
        const int m4 = q << 2;
        for (int j = tid; j < NFFT / 4; j += 256) {
            int grp = j >> (2 * s);
            int pos = j & (q - 1);
            int i0  = grp * m4 + pos;
            int i1 = i0 + q, i2 = i0 + 2 * q, i3 = i0 + 3 * q;
            float2 x0 = a[i0], x1 = a[i1], x2 = a[i2], x3 = a[i3];
            float pr = x0.x + x2.x, pi = x0.y + x2.y;
            float ur = x0.x - x2.x, ui = x0.y - x2.y;
            float qr = x1.x + x3.x, qi = x1.y + x3.y;
            float vr = x1.x - x3.x, vi = x1.y - x3.y;
            float ang = -6.28318530717958647692f * (float)pos / (float)m4;
            float s1, c1; __sincosf(ang, &s1, &c1);
            float c2 = c1 * c1 - s1 * s1, s2 = 2.f * c1 * s1;
            float c3 = c2 * c1 - s2 * s1, s3 = c2 * s1 + s2 * c1;
            a[i0] = make_float2(pr + qr, pi + qi);
            a[i1] = cmul2(make_float2(ur + vi, ui - vr), c1, s1);
            a[i2] = cmul2(make_float2(pr - qr, pi - qi), c2, s2);
            a[i3] = cmul2(make_float2(ur - vi, ui + vr), c3, s3);
        }
        __syncthreads();
    }
}

// ---------------------------------------------------------------------------
// Kernel 2: spectrum of zero-padded filter, scaled by 1/NFFT. One block per d.
// ---------------------------------------------------------------------------
__global__ void __launch_bounds__(256)
fft_filter_kernel(const float* __restrict__ k_in, float2* __restrict__ khat)
{
    extern __shared__ unsigned char dyn_lds[];
    float2* buf = (float2*)dyn_lds;
    const int tid = threadIdx.x;
    const int d   = blockIdx.x;
    const float* kd = k_in + (size_t)d * LSEQ;

    for (int l = tid; l < LSEQ; l += 256) buf[drev14(l)] = make_float2(kd[l], 0.f);
    for (int l = LSEQ + tid; l < NFFT; l += 256) buf[drev14(l)] = make_float2(0.f, 0.f);
    __syncthreads();

    fft4_dit(buf, tid);

    const float scale = 1.0f / (float)NFFT;      // rfft(k)/fft_size folded here
    float2* kh = khat + (size_t)d * KHAT_STRIDE;
    for (int f = tid; f <= LSEQ; f += 256)
        kh[f] = make_float2(buf[f].x * scale, buf[f].y * scale);
}

// ---------------------------------------------------------------------------
// Kernel 3: per-(b,d) channel: FFT(x) * k_hat, inverse via conj+DIF, + x*D[d].
// ---------------------------------------------------------------------------
__global__ void __launch_bounds__(256)
fftconv_kernel(const float* __restrict__ x,
               const float2* __restrict__ khat,
               const float* __restrict__ Dv,
               float* __restrict__ y)
{
    extern __shared__ unsigned char dyn_lds[];
    float2* buf = (float2*)dyn_lds;
    const int tid = threadIdx.x;
    const int d   = blockIdx.x;
    const int b   = blockIdx.y;
    const float* xp = x + ((size_t)b * DMODEL + d) * LSEQ;

    __builtin_prefetch(xp + tid, 0, 0);          // global_prefetch_b8

    for (int l = tid; l < LSEQ; l += 256) buf[drev14(l)] = make_float2(xp[l], 0.f);
    for (int l = LSEQ + tid; l < NFFT; l += 256) buf[drev14(l)] = make_float2(0.f, 0.f);
    __syncthreads();

    fft4_dit(buf, tid);                          // U[f], natural order

    // P = U * K (Hermitian mirror for upper bins); store conj(P) so a forward
    // DIF FFT yields real(IDFT_unnorm(P)) == irfft(..., norm='forward').
    const float2* kh = khat + (size_t)d * KHAT_STRIDE;
    for (int f = tid; f < NFFT; f += 256) {
        float2 K;
        if (f <= LSEQ) { K = kh[f]; }
        else           { K = kh[NFFT - f]; K.y = -K.y; }
        float2 u = buf[f];
        buf[f] = make_float2(u.x * K.x - u.y * K.y,
                             -(u.x * K.y + u.y * K.x));
    }
    __syncthreads();

    fft4_dif(buf, tid);                          // digit-reversed output

    const float dd = Dv[d];
    float* yp = y + ((size_t)b * DMODEL + d) * LSEQ;
    for (int n = tid; n < LSEQ; n += 256)
        yp[n] = buf[drev14(n)].x + xp[n] * dd;
}

// ---------------------------------------------------------------------------
extern "C" void kernel_launch(void* const* d_in, const int* in_sizes, int n_in,
                              void* d_out, int out_size, void* d_ws, size_t ws_size,
                              hipStream_t stream)
{
    (void)in_sizes; (void)n_in; (void)out_size; (void)ws_size;

    const float* x      = (const float*)d_in[0];
    const float* z      = (const float*)d_in[1];
    const float* t      = (const float*)d_in[2];
    const float* freq   = (const float*)d_in[3];
    const float* W1     = (const float*)d_in[4];
    const float* b1     = (const float*)d_in[5];
    const float* W2     = (const float*)d_in[6];
    const float* b2     = (const float*)d_in[7];
    const float* W3     = (const float*)d_in[8];
    const float* b3     = (const float*)d_in[9];
    const float* Wout   = (const float*)d_in[10];
    const float* deltas = (const float*)d_in[11];
    const float* Dv     = (const float*)d_in[12];
    float* out          = (float*)d_out;

    float*  k_ws = (float*)d_ws;                                   // 768*8192 f32
    float2* khat = (float2*)((char*)d_ws + (size_t)DMODEL * LSEQ * sizeof(float));

    // LDS: (3*2*4 + 2*48) B-fragment slots + 4 waves * 2 A-buffers + tpose
    const size_t k1_lds = (size_t)((3 * 2 * 4 + 2 * 48) * BSLOT
                                   + K1_WAVES * 2 * 2 * ASLOT) * sizeof(_Float16)
                        + (size_t)K1_WAVES * 16 * 16 * sizeof(float);

    filter_mlp_kernel<<<dim3(LSEQ / (16 * K1_WAVES)), dim3(K1_THREADS), k1_lds, stream>>>(
        z, t, freq, W1, b1, W2, b2, W3, b3, Wout, deltas, k_ws);

    fft_filter_kernel<<<dim3(DMODEL), dim3(256), NFFT * sizeof(float2), stream>>>(
        k_ws, khat);

    fftconv_kernel<<<dim3(DMODEL, 4), dim3(256), NFFT * sizeof(float2), stream>>>(
        x, khat, Dv, out);
}
